// Attention_2774548873751
// MI455X (gfx1250) — matrix operand order repaired and checked by reading, not compiled, not run on hardware
//
#include <hip/hip_runtime.h>

// ---------------------------------------------------------------------------
// Swin shifted-window attention block for MI455X (gfx1250, wave32, WMMA).
// All matmuls (QKV 1x1 conv, q@k^T, P@v, proj 1x1 conv) use
// v_wmma_f32_16x16x32_bf16 with f32 accumulation; intermediates are bf16 to
// stay on the 23.3 TB/s HBM roofline. Rel-pos bias is folded into the WMMA
// C operand; the shift mask is computed analytically from region ids.
// f32<->bf16 conversions use native __bf16 casts so the compiler can select
// hardware cvt (incl. packed forms) instead of 3-op integer emulation.
// ---------------------------------------------------------------------------

#define NB    8
#define DIM   96
#define HH    256
#define WW    256
#define HW    65536
#define NH    3
#define WS    8
#define SS    4
#define HD    32
#define NTOK  64

typedef __attribute__((ext_vector_type(16))) __bf16 v16bf;
typedef __attribute__((ext_vector_type(8)))  float  v8f;
typedef __attribute__((ext_vector_type(8)))  unsigned short v8us;

union BF16x16 {
  v16bf bf;
  v8us  h[2];
  unsigned short s[16];
};

__device__ __forceinline__ unsigned short f2bf(float f) {
  union { __bf16 b; unsigned short s; } v;
  v.b = (__bf16)f;                     // native cvt, RTNE
  return v.s;
}
__device__ __forceinline__ float bf2f(unsigned short s) {
  union { unsigned short s; __bf16 b; } v;
  v.s = s;
  return (float)v.b;
}
__device__ __forceinline__ v8f wmma_bf16(const BF16x16& a, const BF16x16& b, v8f c) {
  // (neg_a, A, neg_b, B, c_mod, C, reuse_a, reuse_b)
  return __builtin_amdgcn_wmma_f32_16x16x32_bf16(false, a.bf, false, b.bf,
                                                 (short)0, c, false, false);
}

// A-operand K index, contiguous k = h + 16*g (same order as B loads): lane group g, half h.
__device__ __forceinline__ int a_kmap(int h, int g) {
  return h + 16 * g;
}

// ---------------------------------------------------------------------------
// Kernel 1: pack weights into WMMA A-operand order (bf16) + concat biases.
// Apack layout: [(mtile*3 + ktile)*32 + lane]*16 + half
// ---------------------------------------------------------------------------
__global__ void k_pack(const float* __restrict__ V_w,  const float* __restrict__ V_b,
                       const float* __restrict__ QK_w, const float* __restrict__ QK_b,
                       const float* __restrict__ proj_w,
                       unsigned short* __restrict__ Apack1,  // 18*3*32*16
                       float* __restrict__ bias_cat,         // 288
                       unsigned short* __restrict__ Apack2)  // 6*3*32*16
{
  int idx = blockIdx.x * blockDim.x + threadIdx.x;
  if (idx < 18 * 3 * 512) {
    int h    = idx & 15;
    int lane = (idx >> 4) & 31;
    int t    = idx >> 9;          // mt*3 + kt
    int kt   = t % 3;
    int mt   = t / 3;
    int g = lane >> 4, ln = lane & 15;
    int oc = mt * 16 + ln;
    int c  = kt * 32 + a_kmap(h, g);
    float v = (oc < 192) ? QK_w[oc * DIM + c] : V_w[(oc - 192) * DIM + c];
    Apack1[idx] = f2bf(v);
  }
  if (idx < 6 * 3 * 512) {
    int h    = idx & 15;
    int lane = (idx >> 4) & 31;
    int t    = idx >> 9;
    int kt   = t % 3;
    int mt   = t / 3;
    int g = lane >> 4, ln = lane & 15;
    int oc = mt * 16 + ln;
    int c  = kt * 32 + a_kmap(h, g);
    Apack2[idx] = f2bf(proj_w[oc * DIM + c]);
  }
  if (idx < 288) bias_cat[idx] = (idx < 192) ? QK_b[idx] : V_b[idx - 192];
}

// ---------------------------------------------------------------------------
// Kernel 2: rel-pos bias MLP -> bias_g[head][64][64] (f32, query-major)
// ---------------------------------------------------------------------------
__global__ __launch_bounds__(256) void k_bias_mlp(
    const float* __restrict__ w1, const float* __restrict__ b1,
    const float* __restrict__ w2, const float* __restrict__ b2,
    float* __restrict__ bias_g)
{
  int idx = blockIdx.x * 256 + threadIdx.x;       // 0..4095
  if (idx >= NTOK * NTOK) return;
  int n = idx >> 6, m = idx & 63;                 // n = query, m = key
  int i1 = n >> 3, j1 = n & 7, i2 = m >> 3, j2 = m & 7;
  float di = (float)(i1 - i2), dj = (float)(j1 - j2);
  float l0 = log1pf(fabsf(di));
  float l1 = log1pf(fabsf(dj));
  float d0 = (di > 0.f) ? l0 : ((di < 0.f) ? -l0 : 0.f);
  float d1 = (dj > 0.f) ? l1 : ((dj < 0.f) ? -l1 : 0.f);
  float o0 = b2[0], o1 = b2[1], o2 = b2[2];
  for (int k = 0; k < 256; ++k) {
    float hd = fmaxf(0.f, w1[k * 2] * d0 + w1[k * 2 + 1] * d1 + b1[k]);
    o0 += w2[k] * hd;
    o1 += w2[256 + k] * hd;
    o2 += w2[512 + k] * hd;
  }
  bias_g[n * 64 + m]            = o0;
  bias_g[4096 + n * 64 + m]     = o1;
  bias_g[2 * 4096 + n * 64 + m] = o2;
}

// ---------------------------------------------------------------------------
// Kernel 3: fused QK+V 1x1 conv as GEMM. Out channels 0..191 -> QKbuf (bf16),
// 192..287 -> Vbuf (bf16). One wave: 16 out-ch x 64 pixels, K = 96 (3 steps).
// ---------------------------------------------------------------------------
__global__ __launch_bounds__(128) void k_conv1x1_qkv(
    const float* __restrict__ x,                 // [B][96][HW]
    const unsigned short* __restrict__ Apack,    // packed weights
    const float* __restrict__ bias_cat,          // [288]
    unsigned short* __restrict__ QKbuf,          // [B][192][HW]
    unsigned short* __restrict__ Vbuf)           // [B][96][HW]
{
  const int lane = threadIdx.x & 31;
  const int wave = threadIdx.x >> 5;
  const int g = lane >> 4, ln = lane & 15;
  const int mt = blockIdx.y;                     // 0..17
  const int b  = blockIdx.z;
  const int p0 = blockIdx.x * 256 + wave * 64;

  const float* xb = x + (size_t)b * DIM * HW;

  BF16x16 A[3];
#pragma unroll
  for (int kt = 0; kt < 3; ++kt) {
    const v8us* src = (const v8us*)(Apack + (size_t)((mt * 3 + kt) * 32 + lane) * 16);
    A[kt].h[0] = src[0];
    A[kt].h[1] = src[1];
  }

  v8f acc[4];
#pragma unroll
  for (int nt = 0; nt < 4; ++nt) acc[nt] = (v8f){0.f,0.f,0.f,0.f,0.f,0.f,0.f,0.f};

#pragma unroll
  for (int kt = 0; kt < 3; ++kt) {
#pragma unroll
    for (int nt = 0; nt < 4; ++nt) {
      BF16x16 Bt;
      int p = p0 + nt * 16 + ln;
#pragma unroll
      for (int h = 0; h < 16; ++h) {
        int c = kt * 32 + h + 16 * g;            // B layout: K = h + 16*g
        Bt.s[h] = f2bf(xb[(size_t)c * HW + p]);
      }
      acc[nt] = wmma_bf16(A[kt], Bt, acc[nt]);
    }
  }

#pragma unroll
  for (int r = 0; r < 8; ++r) {
    int oc = mt * 16 + r + 8 * g;                // C layout row
    float bv = bias_cat[oc];
#pragma unroll
    for (int nt = 0; nt < 4; ++nt) {
      int p = p0 + nt * 16 + ln;
      unsigned short val = f2bf(acc[nt][r] + bv);
      if (oc < 192) QKbuf[((size_t)b * 192 + oc) * HW + p] = val;
      else          Vbuf[((size_t)b * DIM + (oc - 192)) * HW + p] = val;
    }
  }
}

// ---------------------------------------------------------------------------
// Kernel 4: windowed attention. Block = 96 threads = 3 waves (one per head).
// grid = (1024 windows, 8 batches). Bias enters as WMMA C operand; shift mask
// is analytic from region ids; softmax reduced with shfl_xor over 16 lanes.
// ---------------------------------------------------------------------------
__global__ __launch_bounds__(96) void k_attn(
    const unsigned short* __restrict__ QKbuf,
    const unsigned short* __restrict__ Vbuf,
    const float* __restrict__ bias_g,
    unsigned short* __restrict__ xattn)          // [B][96][HW]
{
  __shared__ __align__(32) unsigned short sQ[NH][NTOK * HD];   // [n][d], q*scale
  __shared__ __align__(32) unsigned short sK[NH][NTOK * HD];   // [n][d]
  __shared__ __align__(32) unsigned short sV[NH][HD * NTOK];   // [d][n] (vT)
  __shared__ __align__(32) unsigned short sP[NH][NTOK * NTOK]; // softmax probs

  const int head = threadIdx.x >> 5;
  const int lane = threadIdx.x & 31;
  const int g = lane >> 4, ln = lane & 15;
  const int wi = blockIdx.x;                     // 0..1023
  const int b  = blockIdx.y;
  const int wy = wi >> 5, wx = wi & 31;
  const float SCALE = 0.17677669529663687f;      // 32^-0.5

  const size_t qbase = ((size_t)b * 192 + head * HD) * HW;
  const size_t kbase = ((size_t)b * 192 + DIM + head * HD) * HW;
  const size_t vbase = ((size_t)b * DIM + head * HD) * HW;

  // Stage q/k/v tiles to LDS (token-major for q/k, d-major for v).
  for (int t = lane; t < NTOK * HD; t += 32) {
    int n = t & 63, d = t >> 6;
    int i = n >> 3, j = n & 7;
    int R = (wy * 8 + i + SS) & 255;
    int C = (wx * 8 + j + SS) & 255;
    int hw = (R << 8) | C;
    unsigned short qv = QKbuf[qbase + (size_t)d * HW + hw];
    sQ[head][n * HD + d] = f2bf(bf2f(qv) * SCALE);
    sK[head][n * HD + d] = QKbuf[kbase + (size_t)d * HW + hw];
    sV[head][d * NTOK + n] = Vbuf[vbase + (size_t)d * HW + hw];
  }
  __syncthreads();

  // Region id of this lane's key columns (n = nt*16 + ln) for the shift mask.
  int colid[4];
#pragma unroll
  for (int nt = 0; nt < 4; ++nt) {
    int n = nt * 16 + ln;
    int Rr = wy * 8 + (n >> 3), Cc = wx * 8 + (n & 7);
    int hg = (Rr < 248) ? 0 : ((Rr < 252) ? 1 : 2);
    int wg = (Cc < 248) ? 0 : ((Cc < 252) ? 1 : 2);
    colid[nt] = hg * 3 + wg;
  }
  const float* biasH = bias_g + head * 4096;

  // Phase 1: S = (q*scale) @ k^T + bias (+mask), row softmax -> sP (bf16).
#pragma unroll
  for (int mt = 0; mt < 4; ++mt) {
    BF16x16 Aq;
    {
      const unsigned short* row = &sQ[head][(mt * 16 + ln) * HD + 16 * g];
      Aq.h[0] = *(const v8us*)row;
      Aq.h[1] = *(const v8us*)(row + 8);
    }
    v8f acc[4];
#pragma unroll
    for (int nt = 0; nt < 4; ++nt) {
      v8f c;
#pragma unroll
      for (int r = 0; r < 8; ++r)
        c[r] = biasH[(mt * 16 + r + 8 * g) * 64 + nt * 16 + ln];
      BF16x16 Bk;
      const unsigned short* row = &sK[head][(nt * 16 + ln) * HD + 16 * g];
      Bk.h[0] = *(const v8us*)row;
      Bk.h[1] = *(const v8us*)(row + 8);
      acc[nt] = wmma_bf16(Aq, Bk, c);
    }
#pragma unroll
    for (int r = 0; r < 8; ++r) {
      int m = mt * 16 + r + 8 * g;                       // query row
      int Rr = wy * 8 + (m >> 3), Cc = wx * 8 + (m & 7);
      int hg = (Rr < 248) ? 0 : ((Rr < 252) ? 1 : 2);
      int wg = (Cc < 248) ? 0 : ((Cc < 252) ? 1 : 2);
      int rid = hg * 3 + wg;
      float s[4];
      float mx = -1e30f;
#pragma unroll
      for (int nt = 0; nt < 4; ++nt) {
        s[nt] = acc[nt][r] + ((colid[nt] == rid) ? 0.f : -100.f);
        mx = fmaxf(mx, s[nt]);
      }
#pragma unroll
      for (int o = 8; o >= 1; o >>= 1) mx = fmaxf(mx, __shfl_xor(mx, o, 32));
      float sum = 0.f;
#pragma unroll
      for (int nt = 0; nt < 4; ++nt) { s[nt] = __expf(s[nt] - mx); sum += s[nt]; }
#pragma unroll
      for (int o = 8; o >= 1; o >>= 1) sum += __shfl_xor(sum, o, 32);
      float inv = 1.0f / sum;
#pragma unroll
      for (int nt = 0; nt < 4; ++nt)
        sP[head][m * 64 + nt * 16 + ln] = f2bf(s[nt] * inv);
    }
  }
  __syncthreads();

  // Phase 2: O = P @ v  (64x64 @ 64x32, K split in two 32-steps).
  v8f out[4][2];
#pragma unroll
  for (int mt = 0; mt < 4; ++mt)
#pragma unroll
    for (int nt = 0; nt < 2; ++nt)
      out[mt][nt] = (v8f){0.f,0.f,0.f,0.f,0.f,0.f,0.f,0.f};

#pragma unroll
  for (int kt = 0; kt < 2; ++kt) {
    BF16x16 Bv[2];
#pragma unroll
    for (int nt = 0; nt < 2; ++nt) {
      const unsigned short* row = &sV[head][(nt * 16 + ln) * NTOK + kt * 32 + 16 * g];
      Bv[nt].h[0] = *(const v8us*)row;
      Bv[nt].h[1] = *(const v8us*)(row + 8);
    }
#pragma unroll
    for (int mt = 0; mt < 4; ++mt) {
      BF16x16 Ap;
      const unsigned short* row = &sP[head][(mt * 16 + ln) * 64 + kt * 32 + 16 * g];
      Ap.h[0] = *(const v8us*)row;
      Ap.h[1] = *(const v8us*)(row + 8);
#pragma unroll
      for (int nt = 0; nt < 2; ++nt)
        out[mt][nt] = wmma_bf16(Ap, Bv[nt], out[mt][nt]);
    }
  }

  // Window-reverse + un-roll store (same +SS mod mapping as the load).
#pragma unroll
  for (int mt = 0; mt < 4; ++mt) {
#pragma unroll
    for (int r = 0; r < 8; ++r) {
      int m = mt * 16 + r + 8 * g;
      int R = (wy * 8 + (m >> 3) + SS) & 255;
      int C = (wx * 8 + (m & 7) + SS) & 255;
      int hw = (R << 8) | C;
#pragma unroll
      for (int nt = 0; nt < 2; ++nt) {
        int ch = head * HD + nt * 16 + ln;
        xattn[((size_t)b * DIM + ch) * HW + hw] = f2bf(out[mt][nt][r]);
      }
    }
  }
}

// ---------------------------------------------------------------------------
// Kernel 5: depthwise 5x5 reflect conv on V + dw bias + attention output.
// ---------------------------------------------------------------------------
__global__ __launch_bounds__(256) void k_dwconv(
    const unsigned short* __restrict__ Vbuf,
    const float* __restrict__ dw_w, const float* __restrict__ dw_b,
    const unsigned short* __restrict__ xattn,
    unsigned short* __restrict__ sum_buf)
{
  int hw = blockIdx.x * 256 + threadIdx.x;
  int c = blockIdx.y, b = blockIdx.z;
  int h = hw >> 8, w = hw & 255;
  const unsigned short* Vc = Vbuf + ((size_t)b * DIM + c) * HW;
  float acc = dw_b[c] + bf2f(xattn[((size_t)b * DIM + c) * HW + hw]);
#pragma unroll
  for (int dy = 0; dy < 5; ++dy) {
    int hh = h + dy - 2;
    hh = (hh < 0) ? -hh : ((hh > 255) ? 510 - hh : hh);
#pragma unroll
    for (int dx = 0; dx < 5; ++dx) {
      int ww = w + dx - 2;
      ww = (ww < 0) ? -ww : ((ww > 255) ? 510 - ww : ww);
      acc += dw_w[c * 25 + dy * 5 + dx] * bf2f(Vc[(hh << 8) | ww]);
    }
  }
  sum_buf[((size_t)b * DIM + c) * HW + hw] = f2bf(acc);
}

// ---------------------------------------------------------------------------
// Kernel 6: output 1x1 conv (proj) GEMM, bf16 in -> f32 out.
// ---------------------------------------------------------------------------
__global__ __launch_bounds__(128) void k_conv1x1_proj(
    const unsigned short* __restrict__ sum_buf,   // [B][96][HW] bf16
    const unsigned short* __restrict__ Apack2,
    const float* __restrict__ proj_b,
    float* __restrict__ out)                      // [B][96][HW] f32
{
  const int lane = threadIdx.x & 31;
  const int wave = threadIdx.x >> 5;
  const int g = lane >> 4, ln = lane & 15;
  const int mt = blockIdx.y;                      // 0..5
  const int b  = blockIdx.z;
  const int p0 = blockIdx.x * 256 + wave * 64;

  const unsigned short* sb = sum_buf + (size_t)b * DIM * HW;

  BF16x16 A[3];
#pragma unroll
  for (int kt = 0; kt < 3; ++kt) {
    const v8us* src = (const v8us*)(Apack2 + (size_t)((mt * 3 + kt) * 32 + lane) * 16);
    A[kt].h[0] = src[0];
    A[kt].h[1] = src[1];
  }

  v8f acc[4];
#pragma unroll
  for (int nt = 0; nt < 4; ++nt) acc[nt] = (v8f){0.f,0.f,0.f,0.f,0.f,0.f,0.f,0.f};

#pragma unroll
  for (int kt = 0; kt < 3; ++kt) {
#pragma unroll
    for (int nt = 0; nt < 4; ++nt) {
      BF16x16 Bt;
      int p = p0 + nt * 16 + ln;
#pragma unroll
      for (int h = 0; h < 16; ++h) {
        int c = kt * 32 + h + 16 * g;
        Bt.s[h] = sb[(size_t)c * HW + p];
      }
      acc[nt] = wmma_bf16(A[kt], Bt, acc[nt]);
    }
  }

#pragma unroll
  for (int r = 0; r < 8; ++r) {
    int oc = mt * 16 + r + 8 * g;
    float bv = proj_b[oc];
#pragma unroll
    for (int nt = 0; nt < 4; ++nt) {
      int p = p0 + nt * 16 + ln;
      out[((size_t)b * DIM + oc) * HW + p] = acc[nt][r] + bv;
    }
  }
}

// ---------------------------------------------------------------------------
// Host launcher
// ---------------------------------------------------------------------------
extern "C" void kernel_launch(void* const* d_in, const int* in_sizes, int n_in,
                              void* d_out, int out_size, void* d_ws, size_t ws_size,
                              hipStream_t stream) {
  (void)in_sizes; (void)n_in; (void)out_size; (void)ws_size;
  const float* x      = (const float*)d_in[0];
  const float* V_w    = (const float*)d_in[1];
  const float* V_b    = (const float*)d_in[2];
  const float* QK_w   = (const float*)d_in[3];
  const float* QK_b   = (const float*)d_in[4];
  const float* proj_w = (const float*)d_in[5];
  const float* proj_b = (const float*)d_in[6];
  const float* dw_w   = (const float*)d_in[7];
  const float* dw_b   = (const float*)d_in[8];
  const float* m_w1   = (const float*)d_in[9];
  const float* m_b1   = (const float*)d_in[10];
  const float* m_w2   = (const float*)d_in[11];
  const float* m_b2   = (const float*)d_in[12];

  char* ws = (char*)d_ws;
  const size_t QK_BYTES = (size_t)NB * 192 * HW * 2;   // 201,326,592
  const size_t V_BYTES  = (size_t)NB * DIM * HW * 2;   // 100,663,296
  unsigned short* QKbuf   = (unsigned short*)(ws);
  unsigned short* Vbuf    = (unsigned short*)(ws + QK_BYTES);
  unsigned short* xattn   = (unsigned short*)(ws + QK_BYTES + V_BYTES);
  // sum_buf overlays QKbuf: QKV fully consumed before the dwconv kernel runs.
  unsigned short* sum_buf = QKbuf;
  char* tail = ws + QK_BYTES + 2 * V_BYTES;
  float*          bias_g   = (float*)(tail);                       // 48 KB
  unsigned short* Apack1   = (unsigned short*)(tail + 49152);      // 54 KB
  float*          bias_cat = (float*)(tail + 49152 + 55296);       // 1.2 KB
  unsigned short* Apack2   = (unsigned short*)(tail + 49152 + 55296 + 1280);

  k_pack<<<(18 * 3 * 512 + 255) / 256, 256, 0, stream>>>(
      V_w, V_b, QK_w, QK_b, proj_w, Apack1, bias_cat, Apack2);

  k_bias_mlp<<<16, 256, 0, stream>>>(m_w1, m_b1, m_w2, m_b2, bias_g);

  k_conv1x1_qkv<<<dim3(HW / 256, 18, NB), 128, 0, stream>>>(
      x, Apack1, bias_cat, QKbuf, Vbuf);

  k_attn<<<dim3(1024, NB), 96, 0, stream>>>(QKbuf, Vbuf, bias_g, xattn);

  k_dwconv<<<dim3(HW / 256, DIM, NB), 256, 0, stream>>>(
      Vbuf, dw_w, dw_b, xattn, sum_buf);

  k_conv1x1_proj<<<dim3(HW / 256, 6, NB), 128, 0, stream>>>(
      sum_buf, Apack2, proj_b, (float*)d_out);
}
